// MPNNNet_74526272520995
// MI455X (gfx1250) — matrix operand order repaired and checked by reading, not compiled, not run on hardware
//
#include <hip/hip_runtime.h>
#include <hip/hip_bf16.h>

// MPNN on MI455X (gfx1250): edge GEMM algebraically split into node GEMMs
// (WMMA bf16, fp32 accum) + L2-resident edge gather/scatter. GEMM kernels are
// fully templated so each launch is branch-free straight-line code.

#define DFEAT 128
#define TK 32
#define LPITCH 40 // bf16 elems per LDS row: 32 data + 8 pad; 80B = 5*16B -> 16B-aligned rows, conflict-free b128

typedef __attribute__((ext_vector_type(16))) __bf16 v16bf;
typedef __attribute__((ext_vector_type(8)))  __bf16 v8bf;
typedef __attribute__((ext_vector_type(4)))  __bf16 v4bf;
typedef __attribute__((ext_vector_type(8)))  float  v8f;

__device__ __forceinline__ __bf16 f2bf(float f) {
  unsigned u = __builtin_bit_cast(unsigned, f);
  u = (u + 0x7fffu + ((u >> 16) & 1u)) >> 16;   // round-to-nearest-even
  unsigned short s = (unsigned short)u;
  return __builtin_bit_cast(__bf16, s);
}

// out[r, 0:128] = act( [in0 | in1*rowscale][r, 0:K] @ W[K,128] + bias ) + resid
// Block: 256 threads (8 waves), 128 rows x 128 cols. Wave w: rows w*16..w*16+15.
template <int K, bool HAS_IN1, bool HAS_BIAS, bool RELU, bool HAS_RESID>
__global__ __launch_bounds__(256)
void gemm128_bf16_wmma(const float* __restrict__ in0,
                       const float* __restrict__ in1,       // K>128 half
                       const float* __restrict__ rowscale,  // applies to in1
                       const float* __restrict__ W,         // [K,128] row-major fp32
                       const float* __restrict__ bias,      // [128]
                       const float* __restrict__ resid,     // [nrows,128]
                       float* __restrict__ out,
                       int nrows) {
  __shared__ __bf16 Xt[128 * LPITCH];   // [row][k]  (k contiguous)
  __shared__ __bf16 Wt[128 * LPITCH];   // [col][k]  (transposed weight tile)

  const int tid  = threadIdx.x;
  const int lane = tid & 31;
  const int wv   = tid >> 5;
  const int half = lane >> 4;
  const int l15  = lane & 15;
  const int rowBase = blockIdx.x * 128;

  v8f acc[8] = {};

#pragma unroll
  for (int k0 = 0; k0 < K; k0 += TK) {
    // ---- stage X tile [128 rows x 32 k] fp32 -> bf16 LDS (coalesced float4 reads)
#pragma unroll
    for (int it = 0; it < 4; ++it) {
      int r  = it * 32 + (tid >> 3);
      int kl = (tid & 7) * 4;
      int gr = min(rowBase + r, nrows - 1);   // clamp: branch-free, pad rows never stored
      int gk = k0 + kl;
      float4 v;
      if (HAS_IN1 && gk >= DFEAT) {           // compile-time-resolvable per k0 when K==128
        v = *(const float4*)(in1 + (size_t)gr * DFEAT + (gk - DFEAT));
        float s = rowscale[gr];
        v.x *= s; v.y *= s; v.z *= s; v.w *= s;
      } else {
        v = *(const float4*)(in0 + (size_t)gr * DFEAT + gk);
      }
      v4bf p;
      p[0] = f2bf(v.x); p[1] = f2bf(v.y); p[2] = f2bf(v.z); p[3] = f2bf(v.w);
      *(v4bf*)(Xt + r * LPITCH + kl) = p;     // ds_store_b64
    }
    // ---- stage W tile [32 k x 128 cols] transposed into Wt[col][k]
#pragma unroll
    for (int it = 0; it < 4; ++it) {
      int kl = (tid >> 5) + it * 8;    // 0..31
      int c  = (tid & 31) * 4;         // 0..124
      float4 w = *(const float4*)(W + (size_t)(k0 + kl) * DFEAT + c);
      Wt[(c + 0) * LPITCH + kl] = f2bf(w.x);
      Wt[(c + 1) * LPITCH + kl] = f2bf(w.y);
      Wt[(c + 2) * LPITCH + kl] = f2bf(w.z);
      Wt[(c + 3) * LPITCH + kl] = f2bf(w.w);
    }
    __syncthreads();

    // ---- A fragment (16x32 bf16): slots 0-7 -> K = half*8+{0..7}; slots 8-15 -> 16+half*8+{0..7}
    const __bf16* ar = Xt + (wv * 16 + l15) * LPITCH;
    v8bf alo = *(const v8bf*)(ar + half * 8);        // ds_load_b128
    v8bf ahi = *(const v8bf*)(ar + 16 + half * 8);   // ds_load_b128
    v16bf afrag = __builtin_shufflevector(alo, ahi,
        0,1,2,3,4,5,6,7,8,9,10,11,12,13,14,15);

    // ---- gather all 8 B fragments first (32x16 bf16: lane = col, K = half*8+{0..7}, 16+half*8+{0..7}),
    //      then issue the wmma chain back-to-back so ds traffic overlaps the matrix pipe.
    v16bf bfrag[8];
#pragma unroll
    for (int t = 0; t < 8; ++t) {
      const __bf16* br = Wt + (t * 16 + l15) * LPITCH;
      v8bf blo = *(const v8bf*)(br + half * 8);
      v8bf bhi = *(const v8bf*)(br + 16 + half * 8);
      bfrag[t] = __builtin_shufflevector(blo, bhi,
          0,1,2,3,4,5,6,7,8,9,10,11,12,13,14,15);
    }
#pragma unroll
    for (int t = 0; t < 8; ++t) {
      acc[t] = __builtin_amdgcn_wmma_f32_16x16x32_bf16(
          false, afrag, false, bfrag[t], (short)0, acc[t], false, false);
    }
    __syncthreads();
  }

  // ---- epilogue: C/D layout VGPR r -> M = r + 8*half, N = lane&15 (+16*t)
#pragma unroll
  for (int t = 0; t < 8; ++t) {
    int col = t * 16 + l15;
    float b = HAS_BIAS ? bias[col] : 0.f;
#pragma unroll
    for (int r = 0; r < 8; ++r) {
      int grow = rowBase + wv * 16 + half * 8 + r;
      if (grow < nrows) {
        float v = acc[t][r] + b;
        if (RELU) v = fmaxf(v, 0.f);
        if (HAS_RESID) v += resid[(size_t)grow * DFEAT + col];
        out[(size_t)grow * DFEAT + col] = v;
      }
    }
  }
}

__global__ void zero_f32(float* __restrict__ p, long n) {
  long i = (long)blockIdx.x * blockDim.x + threadIdx.x;
  long stride = (long)gridDim.x * blockDim.x;
  for (; i < n; i += stride) p[i] = 0.f;
}

__global__ void deg_count(const int* __restrict__ ei, float* __restrict__ deg, int E) {
  int e = blockIdx.x * blockDim.x + threadIdx.x;
  if (e < E) atomicAdd(&deg[ei[E + e]], 1.0f);
}

__global__ void inv_degree(const float* __restrict__ deg, float* __restrict__ invd, int N) {
  int i = blockIdx.x * blockDim.x + threadIdx.x;
  if (i < N) invd[i] = 1.0f / fmaxf(deg[i], 1.0f);
}

// One wave per edge; lane handles 4 floats. m = relu(A[dst]+C[src]+b); agg[dst] += m.
__global__ __launch_bounds__(256)
void edge_message_scatter(const float* __restrict__ A, const float* __restrict__ C,
                          const float* __restrict__ bias, const int* __restrict__ ei,
                          float* __restrict__ agg, int E) {
  int gid = blockIdx.x * blockDim.x + threadIdx.x;
  int e = gid >> 5;
  if (e >= E) return;
  int c   = (gid & 31) * 4;
  int src = ei[e];
  int dst = ei[E + e];
  float4 a = *(const float4*)(A + (size_t)dst * DFEAT + c);
  float4 x = *(const float4*)(C + (size_t)src * DFEAT + c);
  float4 b = *(const float4*)(bias + c);
  float m0 = fmaxf(a.x + x.x + b.x, 0.f);
  float m1 = fmaxf(a.y + x.y + b.y, 0.f);
  float m2 = fmaxf(a.z + x.z + b.z, 0.f);
  float m3 = fmaxf(a.w + x.w + b.w, 0.f);
  float* p = agg + (size_t)dst * DFEAT + c;
  atomicAdd(p + 0, m0);
  atomicAdd(p + 1, m1);
  atomicAdd(p + 2, m2);
  atomicAdd(p + 3, m3);
}

extern "C" void kernel_launch(void* const* d_in, const int* in_sizes, int n_in,
                              void* d_out, int out_size, void* d_ws, size_t ws_size,
                              hipStream_t stream) {
  const float* x     = (const float*)d_in[0];
  const int*   ei    = (const int*)  d_in[1];
  const float* psi_w = (const float*)d_in[2];
  const float* psi_b = (const float*)d_in[3];
  const float* phi_w = (const float*)d_in[4];
  const float* phi_b = (const float*)d_in[5];
  const float* dp_w  = (const float*)d_in[6];
  const float* dp_b  = (const float*)d_in[7];

  const int N = in_sizes[0] / DFEAT;
  const int E = in_sizes[1] / 2;
  const int L = in_sizes[2] / (2 * DFEAT * DFEAT);

  float* ws   = (float*)d_ws;
  float* xA   = ws;                         // [N,128] node state (in-place after layer 0)
  float* bufA = xA   + (size_t)N * DFEAT;   // [N,128] dst-side psi projection
  float* bufC = bufA + (size_t)N * DFEAT;   // [N,128] src-side psi projection
  float* agg  = bufC + (size_t)N * DFEAT;   // [N,128] scatter accumulator
  float* deg  = agg  + (size_t)N * DFEAT;   // [N]
  float* invd = deg  + N;                   // [N]

  const int gemmGrid = (N + 127) / 128;
  const int edgeGrid = (int)(((size_t)E * 32 + 255) / 256);

  // degree / inverse degree (once)
  zero_f32<<<1024, 256, 0, stream>>>(deg, (long)N);
  deg_count<<<(E + 255) / 256, 256, 0, stream>>>(ei, deg, E);
  inv_degree<<<(N + 255) / 256, 256, 0, stream>>>(deg, invd, N);

  const float* cur = x;
  for (int l = 0; l < L; ++l) {
    const float* Wl = psi_w + (size_t)l * 2 * DFEAT * DFEAT;
    // A = x @ psi_w[l][:128,:]  (dst side);  C = x @ psi_w[l][128:,:] (src side)
    gemm128_bf16_wmma<DFEAT, false, false, false, false>
        <<<gemmGrid, 256, 0, stream>>>(
        cur, nullptr, nullptr, Wl, nullptr, nullptr, bufA, N);
    gemm128_bf16_wmma<DFEAT, false, false, false, false>
        <<<gemmGrid, 256, 0, stream>>>(
        cur, nullptr, nullptr, Wl + DFEAT * DFEAT, nullptr, nullptr, bufC, N);

    zero_f32<<<2048, 256, 0, stream>>>(agg, (long)N * DFEAT);
    edge_message_scatter<<<edgeGrid, 256, 0, stream>>>(
        bufA, bufC, psi_b + (size_t)l * DFEAT, ei, agg, E);

    // x = relu([x | agg*inv_deg] @ phi_w[l] + phi_b[l]) + x   (row-local => in-place safe)
    float* nxt = xA;
    gemm128_bf16_wmma<2 * DFEAT, true, true, true, true>
        <<<gemmGrid, 256, 0, stream>>>(
        cur, agg, invd, phi_w + (size_t)l * 2 * DFEAT * DFEAT,
        phi_b + (size_t)l * DFEAT, cur /*residual*/, nxt, N);
    cur = nxt;
  }

  // out = x @ dp_w + dp_b
  gemm128_bf16_wmma<DFEAT, false, true, false, false>
      <<<gemmGrid, 256, 0, stream>>>(
      cur, nullptr, nullptr, dp_w, dp_b, nullptr, (float*)d_out, N);
}
